// BayesianRoutingStrategy_74053826117878
// MI455X (gfx1250) — matrix operand order repaired and checked by reading, not compiled, not run on hardware
//
#include <hip/hip_runtime.h>
#include <hip/hip_bf16.h>
#include <math.h>

typedef __attribute__((ext_vector_type(16))) __bf16 v16bf;
typedef __attribute__((ext_vector_type(8)))  float  v8f;

#define B_TOTAL 16384
#define D_DIM   4096
#define H_DIM   128
#define E_DIM   64
#define S_NUM   5
#define ROWS    64          // rows of B handled per workgroup
#define NTHREADS 128        // 4 wave32s; wave w owns rows 16w..16w+15
#define KSTEPS  (D_DIM / 32)
#define PDROP   0.3f
#define DSCALE  (1.0f / 0.7f)
#define UNC_TH  0.3f

// W1 ws layout: per k-step, one contiguous 16KB block of 4096 dwords:
//   [0..2047]    hi packed-bf16 pairs, [n(128)][kpair(16)]
//   [2048..4095] lo packed-bf16 pairs, same layout
#define WS_BLK_DW 4096

// ---- f32 -> bf16 hi/lo split (native converts, RNE) ---------------------
__device__ __forceinline__ unsigned short bf_bits(__bf16 h) {
  return __builtin_bit_cast(unsigned short, h);
}
__device__ __forceinline__ void split_pack(float a, float b,
                                           unsigned int& hiP, unsigned int& loP) {
  __bf16 ah = (__bf16)a, bh = (__bf16)b;
  __bf16 al = (__bf16)(a - (float)ah);
  __bf16 bl = (__bf16)(b - (float)bh);
  hiP = (unsigned int)bf_bits(ah) | ((unsigned int)bf_bits(bh) << 16);
  loP = (unsigned int)bf_bits(al) | ((unsigned int)bf_bits(bl) << 16);
}

union Frag {
  v16bf v;
  unsigned int u[8];
};

__device__ __forceinline__ v8f wmma_bf16(const Frag& a, const Frag& b, v8f c) {
  return __builtin_amdgcn_wmma_f32_16x16x32_bf16(false, a.v, false, b.v,
                                                 (short)0, c, false, false);
}

// ---- CDNA5 async copy: global -> LDS, 16B per lane, tracked by ASYNCcnt --
__device__ __forceinline__ unsigned lds_off(const void* p) {
  // generic pointer to LDS: low 32 bits are the LDS byte offset
  return (unsigned)(unsigned long long)p;
}
__device__ __forceinline__ void async_copy_b128(unsigned lds_byte, const void* gsrc) {
  asm volatile("global_load_async_to_lds_b128 %0, %1, off"
               :: "v"(lds_byte), "v"(gsrc) : "memory");
}
__device__ __forceinline__ void wait_async0() {
  asm volatile("s_wait_asynccnt 0x0" ::: "memory");
}

// =====================================================================
// Prep: W1 (D x H) -> pre-split, pre-swizzled bf16 hi/lo tiles in ws
// =====================================================================
__global__ __launch_bounds__(256)
void prep_w1(const float* __restrict__ W1, unsigned int* __restrict__ ws) {
  int p = blockIdx.x * 256 + threadIdx.x;       // pair id
  if (p >= KSTEPS * 2048) return;
  int kt  = p >> 11;
  int rem = p & 2047;                           // n*16 + kpair
  int n   = rem >> 4;
  int kp  = rem & 15;
  int k   = kt * 32 + (kp << 1);
  unsigned int hp, lp;
  split_pack(W1[(size_t)k * H_DIM + n], W1[(size_t)(k + 1) * H_DIM + n], hp, lp);
  ws[(size_t)kt * WS_BLK_DW + rem]        = hp;
  ws[(size_t)kt * WS_BLK_DW + 2048 + rem] = lp;
}

// =====================================================================
// Fused main kernel
// =====================================================================
__global__ __launch_bounds__(NTHREADS)
void bayes_route_fused(const float* __restrict__ x,
                       const unsigned int* __restrict__ wsB,   // pre-split W1
                       const float* __restrict__ b1, const float* __restrict__ W2,
                       const float* __restrict__ b2, const float* __restrict__ m1,
                       const float* __restrict__ m2, float* __restrict__ out) {
  // ---- LDS (208 KB of 320 KB) ----
  __shared__ __align__(16) unsigned int Bbuf[2][WS_BLK_DW];   // W1 tile ping-pong (32 KB)
  __shared__ float        hLds[ROWS * H_DIM];                 // relu(x@W1+b1)     (32 KB)
  __shared__ unsigned int w2H[E_DIM * 64], w2L[E_DIM * 64];   // W2^T hi/lo        (32 KB)
  __shared__ unsigned int hsH[ROWS * 64],  hsL[ROWS * 64];    // masked h hi/lo    (32 KB)
  __shared__ float        lgLds[S_NUM * ROWS * E_DIM];        // logits            (80 KB)

  const int tid  = threadIdx.x;
  const int wv   = tid >> 5;     // wave id 0..3
  const int lane = tid & 31;
  const int lr   = lane & 15;
  const int hh   = lane >> 4;    // lane half
  const int b0   = blockIdx.x * ROWS;

  // ---- stage W2 -> LDS bf16 hi/lo, [e][kpair] (once per WG, 32 pairs/thread)
  for (int p = tid; p < E_DIM * 64; p += NTHREADS) {
    int e  = p >> 6;
    int k  = (p & 63) << 1;
    unsigned int hp, lp;
    split_pack(W2[(size_t)k * E_DIM + e], W2[(size_t)(k + 1) * E_DIM + e], hp, lp);
    w2H[p] = hp;
    w2L[p] = lp;
  }

  const unsigned bb0 = lds_off(&Bbuf[0][0]) + tid * 128;
  const unsigned bb1 = lds_off(&Bbuf[1][0]) + tid * 128;

  // prologue: async-copy W1 block 0 into Bbuf[0] (128 B per thread)
  {
    const unsigned int* src = wsB + tid * 32;
#pragma unroll
    for (int i = 0; i < 8; ++i)
      async_copy_b128(bb0 + 16 * i, src + 4 * i);
    wait_async0();
  }
  __syncthreads();

  // A pipeline: two contiguous 32B chunks of this lane's row per k-step
  const float* xrow = x + (size_t)(b0 + 16 * wv + lr) * D_DIM;
  float4 a0 = *(const float4*)(xrow + 8 * hh);
  float4 a1 = *(const float4*)(xrow + 8 * hh + 4);
  float4 a2 = *(const float4*)(xrow + 16 + 8 * hh);
  float4 a3 = *(const float4*)(xrow + 16 + 8 * hh + 4);

  v8f acc[8] = {};
  for (int kt = 0; kt < KSTEPS; ++kt) {
    const unsigned int* cur = &Bbuf[kt & 1][0];

    // async-copy next W1 block into the other buffer (overlaps the WMMAs)
    if (kt + 1 < KSTEPS) {
      const unsigned int* src = wsB + (size_t)(kt + 1) * WS_BLK_DW + tid * 32;
      unsigned dst = (kt & 1) ? bb0 : bb1;
#pragma unroll
      for (int i = 0; i < 8; ++i)
        async_copy_b128(dst + 16 * i, src + 4 * i);
    }

    // convert current A chunk to hi/lo fragments (ISA 16x32 bf16 A layout)
    Frag aH, aL;
    split_pack(a0.x, a0.y, aH.u[0], aL.u[0]);
    split_pack(a0.z, a0.w, aH.u[1], aL.u[1]);
    split_pack(a1.x, a1.y, aH.u[2], aL.u[2]);
    split_pack(a1.z, a1.w, aH.u[3], aL.u[3]);
    split_pack(a2.x, a2.y, aH.u[4], aL.u[4]);
    split_pack(a2.z, a2.w, aH.u[5], aL.u[5]);
    split_pack(a3.x, a3.y, aH.u[6], aL.u[6]);
    split_pack(a3.z, a3.w, aH.u[7], aL.u[7]);

    // prefetch next A chunk
    if (kt + 1 < KSTEPS) {
      const float* nx = xrow + (size_t)(kt + 1) * 32;
      a0 = *(const float4*)(nx + 8 * hh);
      a1 = *(const float4*)(nx + 8 * hh + 4);
      a2 = *(const float4*)(nx + 16 + 8 * hh);
      a3 = *(const float4*)(nx + 16 + 8 * hh + 4);
    }

#pragma unroll
    for (int t = 0; t < 8; ++t) {
      Frag bH, bL;
      int n = 16 * t + lr;
#pragma unroll
      for (int v = 0; v < 8; ++v) {
        int idx = n * 16 + ((v < 4) ? (4 * hh + v) : (8 + 4 * hh + (v - 4)));     // K = 8*hh + 2v (v<4), 16 + 8*hh + 2(v-4) (v>=4)
        bH.u[v] = cur[idx];
        bL.u[v] = cur[2048 + idx];
      }
      acc[t] = wmma_bf16(aH, bH, acc[t]);
      acc[t] = wmma_bf16(aH, bL, acc[t]);
      acc[t] = wmma_bf16(aL, bH, acc[t]);
    }

    wait_async0();       // next buffer fully in LDS before the barrier
    __syncthreads();
  }

  // bias + relu -> LDS h tile.  C layout: VGPR v -> M = v + 8*half, N = lane%16
#pragma unroll
  for (int t = 0; t < 8; ++t) {
    int col = 16 * t + lr;
    float bias = b1[col];
#pragma unroll
    for (int v = 0; v < 8; ++v) {
      int row = 16 * wv + v + 8 * hh;
      hLds[row * H_DIM + col] = fmaxf(acc[t][v] + bias, 0.0f);
    }
  }
  __syncthreads();

  // =====================================================================
  // Phase 2: per-sample dropout + GEMM2 (K = 128) + bias + mask2 -> logits
  // =====================================================================
  for (int s = 0; s < S_NUM; ++s) {
    const float* m1s = m1 + (size_t)s * B_TOTAL * H_DIM + (size_t)b0 * H_DIM;
    for (int p = tid; p < ROWS * 64; p += NTHREADS) {
      int r = p >> 6;
      int k = (p & 63) << 1;
      float ka = (m1s[r * H_DIM + k]     >= PDROP) ? DSCALE : 0.0f;
      float kb = (m1s[r * H_DIM + k + 1] >= PDROP) ? DSCALE : 0.0f;
      unsigned int hp, lp;
      split_pack(hLds[r * H_DIM + k] * ka, hLds[r * H_DIM + k + 1] * kb, hp, lp);
      hsH[p] = hp;
      hsL[p] = lp;
    }
    __syncthreads();

    v8f lacc[4] = {};
#pragma unroll
    for (int kt = 0; kt < 4; ++kt) {
      int kb0 = kt * 32;
      Frag aH, aL;
#pragma unroll
      for (int v = 0; v < 8; ++v) {
        int K   = kb0 + 16 * hh + 2 * v;
        int idx = (16 * wv + lr) * 64 + (K >> 1);
        aH.u[v] = hsH[idx];
        aL.u[v] = hsL[idx];
      }
#pragma unroll
      for (int t = 0; t < 4; ++t) {
        Frag bH, bL;
        int e = 16 * t + lr;
#pragma unroll
        for (int v = 0; v < 8; ++v) {
          int idx = e * 64 + ((kb0 + 16 * hh + 2 * v) >> 1);
          bH.u[v] = w2H[idx];
          bL.u[v] = w2L[idx];
        }
        lacc[t] = wmma_bf16(aH, bH, lacc[t]);
        lacc[t] = wmma_bf16(aH, bL, lacc[t]);
        lacc[t] = wmma_bf16(aL, bH, lacc[t]);
      }
    }
    const float* m2s = m2 + (size_t)s * B_TOTAL * E_DIM;
#pragma unroll
    for (int t = 0; t < 4; ++t) {
      int col = 16 * t + lr;
      float bias = b2[col];
#pragma unroll
      for (int v = 0; v < 8; ++v) {
        int row = 16 * wv + v + 8 * hh;
        int gb  = b0 + row;
        float keep = (m2s[(size_t)gb * E_DIM + col] >= PDROP) ? DSCALE : 0.0f;
        lgLds[(s * ROWS + row) * E_DIM + col] = (lacc[t][v] + bias) * keep;
      }
    }
    __syncthreads();
  }

  // =====================================================================
  // Phase 3: per-row softmax stats, uncertainty, top-4 (1 thread / row)
  // =====================================================================
  float* meanBuf = (float*)hsH;   // reuse (hs dead now)
  if (tid < ROWS) {
    const int row = tid;
    const int gb  = b0 + row;

    float rmax[S_NUM], rsum[S_NUM];
    for (int s = 0; s < S_NUM; ++s) {
      const float* L = &lgLds[(s * ROWS + row) * E_DIM];
      float m = -3.4e38f;
      for (int e = 0; e < E_DIM; ++e) m = fmaxf(m, L[e]);
      float sum = 0.0f;
      for (int e = 0; e < E_DIM; ++e) sum += expf(L[e] - m);
      rmax[s] = m;
      rsum[s] = sum;
    }

    float unc = 0.0f, mmax = -3.4e38f;
    for (int e = 0; e < E_DIM; ++e) {
      float msum = 0.0f, ps = 0.0f, pss = 0.0f;
      for (int s = 0; s < S_NUM; ++s) {
        float l = lgLds[(s * ROWS + row) * E_DIM + e];
        msum += l;
        float p = expf(l - rmax[s]) / rsum[s];
        ps  += p;
        pss += p * p;
      }
      float mean = msum * 0.2f;
      meanBuf[row * E_DIM + e] = mean;
      mmax = fmaxf(mmax, mean);
      float var = (pss - ps * ps * 0.2f) * 0.25f;   // ddof=1
      unc += sqrtf(fmaxf(var, 0.0f));
    }
    unc *= (1.0f / E_DIM);

    float se = 0.0f;
    for (int e = 0; e < E_DIM; ++e) se += expf(meanBuf[row * E_DIM + e] - mmax);

    int   sel[4];
    float selp[4];
    for (int j = 0; j < 4; ++j) {
      float best = -3.4e38f;
      int   bi   = 0;
      for (int e = 0; e < E_DIM; ++e) {
        bool used = false;
        for (int q = 0; q < j; ++q) used = used || (sel[q] == e);
        float v = meanBuf[row * E_DIM + e];
        if (!used && v > best) { best = v; bi = e; }
      }
      sel[j]  = bi;
      selp[j] = expf(best - mmax) / se;
    }

    bool hi = unc > UNC_TH;
    float* oIdx = out;
    float* oPrb = out + (size_t)B_TOTAL * 4;
    float* oUnc = out + (size_t)B_TOTAL * 8;
    oIdx[gb * 4 + 0] = (float)sel[0];
    oIdx[gb * 4 + 1] = (float)sel[1];
    oIdx[gb * 4 + 2] = hi ? (float)sel[2] : -1.0f;
    oIdx[gb * 4 + 3] = hi ? (float)sel[3] : -1.0f;
    oPrb[gb * 4 + 0] = selp[0];
    oPrb[gb * 4 + 1] = selp[1];
    oPrb[gb * 4 + 2] = hi ? selp[2] : 0.0f;
    oPrb[gb * 4 + 3] = hi ? selp[3] : 0.0f;
    oUnc[gb] = unc;
  }
}

extern "C" void kernel_launch(void* const* d_in, const int* in_sizes, int n_in,
                              void* d_out, int out_size, void* d_ws, size_t ws_size,
                              hipStream_t stream) {
  (void)in_sizes; (void)n_in; (void)out_size; (void)ws_size;
  const float* x  = (const float*)d_in[0];
  const float* W1 = (const float*)d_in[1];
  const float* b1 = (const float*)d_in[2];
  const float* W2 = (const float*)d_in[3];
  const float* b2 = (const float*)d_in[4];
  const float* m1 = (const float*)d_in[5];
  const float* m2 = (const float*)d_in[6];
  float* out = (float*)d_out;
  unsigned int* ws = (unsigned int*)d_ws;   // needs 2 MB: pre-split W1 tiles

  hipLaunchKernelGGL(prep_w1, dim3((KSTEPS * 2048 + 255) / 256), dim3(256),
                     0, stream, W1, ws);
  hipLaunchKernelGGL(bayes_route_fused, dim3(B_TOTAL / ROWS), dim3(NTHREADS),
                     0, stream, x, ws, b1, W2, b2, m1, m2, out);
}
